// SuperBEVGlue_22368189678236
// MI455X (gfx1250) — matrix operand order repaired and checked by reading, not compiled, not run on hardware
//
#include <hip/hip_runtime.h>
#include <hip/hip_bf16.h>
#include <math.h>

typedef _Float16 f16;
typedef __attribute__((ext_vector_type(16))) _Float16 v16h;
typedef __attribute__((ext_vector_type(8)))  _Float16 v8h;
typedef __attribute__((ext_vector_type(4)))  _Float16 v4h;
typedef __attribute__((ext_vector_type(8)))  float    v8f;
typedef __attribute__((ext_vector_type(4)))  float    v4f;

#define TILE_M 64
#define TILE_N 128
#define TILE_K 32
#define LDA_SH 40    /* TILE_K + 8 halves */
#define LDB_SH 136   /* TILE_N + 8 halves */

// ---------------------------------------------------------------------------
// Fused-epilogue parameters for the generic WMMA GEMM.
// y = acc*alpha (+bias[m]) (-> BN fold) (+res) (relu) -> out16/out32 (strided)
// ---------------------------------------------------------------------------
struct Ep {
  const float* bias = nullptr;
  const float* bn_g = nullptr; const float* bn_b = nullptr;
  const float* bn_m = nullptr; const float* bn_v = nullptr;
  int   relu  = 0;
  float alpha = 1.0f;
  const f16*   res16 = nullptr; long r16_ms = 0, r16_ns = 0;
  const float* res32 = nullptr; long r32_ms = 0, r32_ns = 0;
  f16*   out16 = nullptr; long o16_ms = 0, o16_ns = 0;
  float* out32 = nullptr; long o32_ms = 0, o32_ns = 0;
};

// ---------------------------------------------------------------------------
// Generic strided GEMM:  C[M][N] = alpha * sum_k A[m][k]*B[k][n]  (+ epilogue)
// A element at A + m*a_rs + k*a_ks ; B element at B + k*b_ks + n*b_ns.
// TA/TB = float or _Float16; converted to f16 while staging to LDS.
// Block: 256 threads = 8 waves (wave32), tile 64x128, K-step 32.
// Each wave owns a 32x32 sub-tile = 2x2 v_wmma_f32_16x16x32_f16.
// Staging uses uniform-branch vector fast paths (b128) and, in the generic
// fallback, clamped unconditional loads gathered to registers first so the
// loads pipeline with a single wait (no per-element exec branching).
// ---------------------------------------------------------------------------
template <typename TA, typename TB>
__global__ __launch_bounds__(256) void wmma_gemm_kernel(
    const TA* __restrict__ A, long a_rs, long a_ks,
    const TB* __restrict__ B, long b_ks, long b_ns,
    int M, int N, int K, Ep ep)
{
  __shared__ alignas(16) f16 As[TILE_M * LDA_SH];
  __shared__ alignas(16) f16 Bs[TILE_K * LDB_SH];

  const int tid  = threadIdx.x;
  const int lane = tid & 31;
  const int wave = tid >> 5;
  const int wm   = wave >> 2;   // 0..1
  const int wn   = wave & 3;    // 0..3
  const int bm   = blockIdx.y * TILE_M;
  const int bn   = blockIdx.x * TILE_N;
  const bool fullM = (bm + TILE_M <= M);
  const bool fullN = (bn + TILE_N <= N);

  v8f acc[2][2];
  for (int i = 0; i < 2; ++i)
    for (int j = 0; j < 2; ++j)
      for (int g = 0; g < 8; ++g) acc[i][j][g] = 0.0f;

  for (int k0 = 0; k0 < K; k0 += TILE_K) {
    const bool fullK = (k0 + TILE_K <= K);

    // ================= stage A tile (64 x 32) =================
    bool doneA = false;
    if (fullM && fullK) {
      if constexpr (sizeof(TA) == 2) {
        if (a_ks == 1) {
          // contiguous along k: one b128 load + one b128 LDS store per thread
          int m = tid >> 2, c = tid & 3;
          v8h vv = *(const v8h*)(A + (long)(bm + m) * a_rs + (long)(k0 + c * 8));
          *(v8h*)&As[m * LDA_SH + c * 8] = vv;
          doneA = true;
        } else if (a_rs == 1) {
          // contiguous along m: coalesced b128 loads, strided b16 LDS stores
          int c = tid & 7, k = tid >> 3;
          v8h vv = *(const v8h*)(A + (long)(bm + c * 8) + (long)(k0 + k) * a_ks);
#pragma unroll
          for (int j = 0; j < 8; ++j) As[(c * 8 + j) * LDA_SH + k] = vv[j];
          doneA = true;
        }
      } else {
        if (a_ks == 1) {
          // fp32 weights, contiguous along k: float4 loads + packed cvt
#pragma unroll
          for (int t = 0; t < 2; ++t) {
            int slot = tid + t * 256;
            int m = slot >> 3, c = slot & 7;
            v4f f = *(const v4f*)(A + (long)(bm + m) * a_rs + (long)(k0 + c * 4));
            v4h h;
#pragma unroll
            for (int j = 0; j < 4; ++j) h[j] = (f16)f[j];
            *(v4h*)&As[m * LDA_SH + c * 4] = h;
          }
          doneA = true;
        }
      }
    }
    if (!doneA) {
      // generic: clamped unconditional loads -> registers, then convert/store
      TA tmpA[8]; bool okA[8];
#pragma unroll
      for (int t = 0; t < 8; ++t) {
        int idx = tid + t * 256;
        int m = idx >> 5, k = idx & 31;
        okA[t] = (bm + m < M) && (k0 + k < K);
        long o = okA[t] ? ((long)(bm + m) * a_rs + (long)(k0 + k) * a_ks) : 0;
        tmpA[t] = A[o];
      }
#pragma unroll
      for (int t = 0; t < 8; ++t) {
        int idx = tid + t * 256;
        int m = idx >> 5, k = idx & 31;
        As[m * LDA_SH + k] = okA[t] ? (f16)(float)tmpA[t] : (f16)0.f;
      }
    }

    // ================= stage B tile (32 x 128) =================
    bool doneB = false;
    if (fullN && fullK) {
      if constexpr (sizeof(TB) == 2) {
        if (b_ns == 1) {
          // contiguous along n: b128 load -> b128 LDS store (2 per thread)
#pragma unroll
          for (int t = 0; t < 2; ++t) {
            int slot = tid + t * 256;
            int k = slot >> 4, c = slot & 15;
            v8h vv = *(const v8h*)(B + (long)(k0 + k) * b_ks + (long)(bn + c * 8));
            *(v8h*)&Bs[k * LDB_SH + c * 8] = vv;
          }
          doneB = true;
        }
      } else {
        if (b_ks == 1) {
          // fp32 softmax matrix, contiguous along k: float4 loads + cvt
#pragma unroll
          for (int t = 0; t < 4; ++t) {
            int slot = tid + t * 256;
            int n = slot >> 3, c = slot & 7;
            v4f f = *(const v4f*)(B + (long)(k0 + c * 4) + (long)(bn + n) * b_ns);
#pragma unroll
            for (int j = 0; j < 4; ++j) Bs[(c * 4 + j) * LDB_SH + n] = (f16)f[j];
          }
          doneB = true;
        }
      }
    }
    if (!doneB) {
      TB tmpB[16]; bool okB[16];
#pragma unroll
      for (int t = 0; t < 16; ++t) {
        int idx = tid + t * 256;
        int n = idx & 127, k = idx >> 7;
        okB[t] = (k0 + k < K) && (bn + n < N);
        long o = okB[t] ? ((long)(k0 + k) * b_ks + (long)(bn + n) * b_ns) : 0;
        tmpB[t] = B[o];
      }
#pragma unroll
      for (int t = 0; t < 16; ++t) {
        int idx = tid + t * 256;
        int n = idx & 127, k = idx >> 7;
        Bs[k * LDB_SH + n] = okB[t] ? (f16)(float)tmpB[t] : (f16)0.f;
      }
    }
    __syncthreads();

    // ---- fragments ----
    // A (16x32): lanes 0-15 row M=r (K 0..7,16..23), lanes 16-31 same row (K 8..15,24..31)
    const int r  = lane & 15;
    const int hi = lane >> 4;
    v16h af[2], bf[2];
#pragma unroll
    for (int im = 0; im < 2; ++im) {
      const f16* src = &As[(wm * 32 + im * 16 + r) * LDA_SH + hi * 8];
      v8h lo = *(const v8h*)src;
      v8h hh = *(const v8h*)(src + 16);
#pragma unroll
      for (int e = 0; e < 8; ++e) { af[im][e] = lo[e]; af[im][e + 8] = hh[e]; }
    }
    // B (32x16): lanes 0-15 col N=r (K 0..7,16..23), lanes 16-31 same col (K 8..15,24..31)
#pragma unroll
    for (int in = 0; in < 2; ++in) {
      const f16* src = &Bs[(hi * 8) * LDB_SH + wn * 32 + in * 16 + r];
#pragma unroll
      for (int e = 0; e < 8; ++e) { bf[in][e] = src[e * LDB_SH]; bf[in][e + 8] = src[(e + 16) * LDB_SH]; }
    }
#pragma unroll
    for (int im = 0; im < 2; ++im)
#pragma unroll
      for (int in = 0; in < 2; ++in)
        acc[im][in] = __builtin_amdgcn_wmma_f32_16x16x32_f16(
            false, af[im], false, bf[in], (short)0, acc[im][in], false, false);
    __syncthreads();
  }

  // ---- epilogue (C layout: VGPR g -> M = g + 8*hi ; N = lane&15) ----
  const int r  = lane & 15;
  const int hi = lane >> 4;
#pragma unroll
  for (int im = 0; im < 2; ++im) {
#pragma unroll
    for (int in = 0; in < 2; ++in) {
      int n = bn + wn * 32 + in * 16 + r;
#pragma unroll
      for (int g = 0; g < 8; ++g) {
        int m = bm + wm * 32 + im * 16 + g + hi * 8;
        if (m < M && n < N) {
          float y = acc[im][in][g] * ep.alpha;
          if (ep.bias) y += ep.bias[m];
          if (ep.bn_g) {
            float s = ep.bn_g[m] * __frsqrt_rn(ep.bn_v[m] + 1e-5f);
            y = y * s + (ep.bn_b[m] - ep.bn_m[m] * s);
          }
          if (ep.res32) y += ep.res32[(long)m * ep.r32_ms + (long)n * ep.r32_ns];
          if (ep.res16) y += (float)ep.res16[(long)m * ep.r16_ms + (long)n * ep.r16_ns];
          if (ep.relu && y < 0.f) y = 0.f;
          if (ep.out16) ep.out16[(long)m * ep.o16_ms + (long)n * ep.o16_ns] = (f16)y;
          if (ep.out32) ep.out32[(long)m * ep.o32_ms + (long)n * ep.o32_ns] = y;
        }
      }
    }
  }
}

template <typename TA, typename TB>
static inline void launch_gemm(hipStream_t s, const TA* A, long ars, long aks,
                               const TB* B, long bks, long bns,
                               int M, int N, int K, const Ep& ep) {
  dim3 grid((N + TILE_N - 1) / TILE_N, (M + TILE_M - 1) / TILE_M);
  wmma_gemm_kernel<TA, TB><<<grid, 256, 0, s>>>(A, ars, aks, B, bks, bns, M, N, K, ep);
}

// ---------------------------------------------------------------------------
// Elementwise / reduction kernels
// ---------------------------------------------------------------------------
__global__ void build_e_kernel(const float* __restrict__ kpts,
                               const float* __restrict__ sc,
                               f16* __restrict__ e, int N) {
  int n = blockIdx.x * 256 + threadIdx.x;
  if (n >= N) return;
  const float inv = 1.f / (352.f * 0.7f);
  e[0 * N + n] = (f16)((kpts[n * 4 + 0] - 176.f) * inv);
  e[1 * N + n] = (f16)((kpts[n * 4 + 1] - 176.f) * inv);
  e[2 * N + n] = (f16)((kpts[n * 4 + 2] - 50.f) * inv);
  e[3 * N + n] = (f16)((kpts[n * 4 + 3] - 50.f) * inv);
  e[4 * N + n] = (f16)sc[n];
}

__global__ void resadd_kernel(f16* __restrict__ d, const f16* __restrict__ delta, int n) {
  int i = blockIdx.x * 256 + threadIdx.x;
  if (i < n) d[i] = (f16)((float)d[i] + (float)delta[i]);
}

__global__ __launch_bounds__(256) void row_softmax_kernel(float* __restrict__ S, int cols) {
  float* p = S + (long)blockIdx.x * cols;
  __shared__ float red[256];
  const int tid = threadIdx.x;
  float vals[8];
  int cnt = 0;
  float mx = -3.4e38f;
  for (int j = tid; j < cols; j += 256) { float a = p[j]; vals[cnt++] = a; mx = fmaxf(mx, a); }
  red[tid] = mx; __syncthreads();
  for (int o = 128; o > 0; o >>= 1) { if (tid < o) red[tid] = fmaxf(red[tid], red[tid + o]); __syncthreads(); }
  mx = red[0]; __syncthreads();
  float sum = 0.f;
  for (int i = 0; i < cnt; ++i) { vals[i] = __expf(vals[i] - mx); sum += vals[i]; }
  red[tid] = sum; __syncthreads();
  for (int o = 128; o > 0; o >>= 1) { if (tid < o) red[tid] += red[tid + o]; __syncthreads(); }
  float inv = 1.f / red[0];
  cnt = 0;
  for (int j = tid; j < cols; j += 256) p[j] = vals[cnt++] * inv;
}

__global__ void fill_bins_kernel(float* __restrict__ C, int n, const float* __restrict__ bin) {
  int i = blockIdx.x * 256 + threadIdx.x;
  if (i >= n) return;
  float a = *bin;
  C[(long)i * n + (n - 1)] = a;
  C[(long)(n - 1) * n + i] = a;
}

__global__ void zero_uv_kernel(float* u, float* v, int n) {
  int i = blockIdx.x * 256 + threadIdx.x;
  if (i < n) { u[i] = 0.f; v[i] = 0.f; }
}

// u[i] = log_mu[i] - logsumexp_j( C[i][j] + v[j] ), one block per row
__global__ __launch_bounds__(256) void sink_row_kernel(
    const float* __restrict__ C, const float* __restrict__ v, float* __restrict__ u,
    int n, float lmu, float lmu_last) {
  const float* p = C + (long)blockIdx.x * n;
  __shared__ float rm[256], rs[256];
  const int tid = threadIdx.x;
  float m = -3.4e38f, s = 0.f;
  for (int j = tid; j < n; j += 256) {
    float a = p[j] + v[j];
    if (a > m) { s = s * __expf(m - a) + 1.f; m = a; }
    else       { s += __expf(a - m); }
  }
  rm[tid] = m; rs[tid] = s; __syncthreads();
  for (int o = 128; o > 0; o >>= 1) {
    if (tid < o) {
      float m2 = rm[tid + o], s2 = rs[tid + o];
      float mm = fmaxf(rm[tid], m2);
      rs[tid] = rs[tid] * __expf(rm[tid] - mm) + s2 * __expf(m2 - mm);
      rm[tid] = mm;
    }
    __syncthreads();
  }
  if (tid == 0) {
    float lse = rm[0] + __logf(rs[0]);
    u[blockIdx.x] = ((blockIdx.x == (unsigned)(n - 1)) ? lmu_last : lmu) - lse;
  }
}

// Column logsumexp pass 1: thread-per-column (coalesced), 16 row-slices.
__global__ __launch_bounds__(256) void sink_col_part_kernel(
    const float* __restrict__ C, const float* __restrict__ u,
    float* __restrict__ pm, float* __restrict__ ps, int n, int chunk) {
  int j = blockIdx.x * 256 + threadIdx.x;
  if (j >= n) return;
  int r0 = blockIdx.y * chunk;
  int r1 = (r0 + chunk < n) ? (r0 + chunk) : n;
  float m = -3.4e38f, s = 0.f;
  for (int i = r0; i < r1; ++i) {
    float a = C[(long)i * n + j] + u[i];
    if (a > m) { s = s * __expf(m - a) + 1.f; m = a; }
    else       { s += __expf(a - m); }
  }
  pm[(long)blockIdx.y * n + j] = m;
  ps[(long)blockIdx.y * n + j] = s;
}

__global__ void sink_col_comb_kernel(const float* __restrict__ pm, const float* __restrict__ ps,
                                     float* __restrict__ v, int n, int nslice,
                                     float lnu, float lnu_last) {
  int j = blockIdx.x * 256 + threadIdx.x;
  if (j >= n) return;
  float m = -3.4e38f, s = 0.f;
  for (int t = 0; t < nslice; ++t) {
    float m2 = pm[(long)t * n + j], s2 = ps[(long)t * n + j];
    float mm = fmaxf(m, m2);
    s = s * __expf(m - mm) + s2 * __expf(m2 - mm);
    m = mm;
  }
  v[j] = ((j == n - 1) ? lnu_last : lnu) - (m + __logf(s));
}

__global__ void write_z_kernel(const float* __restrict__ C, const float* __restrict__ u,
                               const float* __restrict__ v, float* __restrict__ Z,
                               int n, float negnorm) {
  long idx = (long)blockIdx.x * 256 + threadIdx.x;
  long total = (long)n * n;
  if (idx >= total) return;
  int i = (int)(idx / n), j = (int)(idx % n);
  Z[idx] = C[idx] + u[i] + v[j] + negnorm;
}

__global__ __launch_bounds__(256) void row_argmax_kernel(
    const float* __restrict__ Z, int ld, int cols, float* __restrict__ mx, int* __restrict__ ix) {
  const float* p = Z + (long)blockIdx.x * ld;
  __shared__ float rv[256];
  __shared__ int   ri[256];
  int tid = threadIdx.x;
  float best = -3.4e38f; int bj = 0;
  for (int j = tid; j < cols; j += 256) {
    float a = p[j];
    if (a > best) { best = a; bj = j; }
  }
  rv[tid] = best; ri[tid] = bj; __syncthreads();
  for (int o = 128; o > 0; o >>= 1) {
    if (tid < o) {
      if (rv[tid + o] > rv[tid] || (rv[tid + o] == rv[tid] && ri[tid + o] < ri[tid])) {
        rv[tid] = rv[tid + o]; ri[tid] = ri[tid + o];
      }
    }
    __syncthreads();
  }
  if (tid == 0) { mx[blockIdx.x] = rv[0]; ix[blockIdx.x] = ri[0]; }
}

__global__ void col_argmax_kernel(const float* __restrict__ Z, int ld, int rows, int cols,
                                  int* __restrict__ ix) {
  int j = blockIdx.x * 256 + threadIdx.x;
  if (j >= cols) return;
  float best = -3.4e38f; int bi = 0;
  for (int i = 0; i < rows; ++i) {
    float a = Z[(long)i * ld + j];
    if (a > best) { best = a; bi = i; }
  }
  ix[j] = bi;
}

__global__ void match0_kernel(const float* __restrict__ mx0, const int* __restrict__ ix0,
                              const int* __restrict__ ix1, float* __restrict__ out_idx0,
                              float* __restrict__ out_msc0, int* __restrict__ valid0, int N) {
  int i = blockIdx.x * 256 + threadIdx.x;
  if (i >= N) return;
  int j = ix0[i];
  bool mutual = (ix1[j] == i);
  float m = mutual ? __expf(mx0[i]) : 0.f;
  bool valid = mutual && (m > 0.2f);
  out_msc0[i] = m;
  valid0[i]   = valid ? 1 : 0;
  out_idx0[i] = valid ? (float)j : -1.f;
}

__global__ void match1_kernel(const int* __restrict__ ix0, const int* __restrict__ ix1,
                              const float* __restrict__ msc0, const int* __restrict__ valid0,
                              float* __restrict__ out_idx1, float* __restrict__ out_msc1, int N) {
  int j = blockIdx.x * 256 + threadIdx.x;
  if (j >= N) return;
  int i = ix1[j];
  bool mutual = (ix0[i] == j);
  out_msc1[j] = mutual ? msc0[i] : 0.f;
  bool valid = mutual && (valid0[i] != 0);
  out_idx1[j] = valid ? (float)i : -1.f;
}

// ---------------------------------------------------------------------------
// Host orchestration
// ---------------------------------------------------------------------------
extern "C" void kernel_launch(void* const* d_in, const int* in_sizes, int n_in,
                              void* d_out, int out_size, void* d_ws, size_t ws_size,
                              hipStream_t stream) {
  (void)in_sizes; (void)n_in; (void)out_size; (void)ws_size;
  constexpr int N = 2048, D = 256, NB = 2049, H = 4, HD = 64, H2 = 512, LAYERS = 18;

  // ---- input mapping (top level in dict order; params jax-tree sorted) ----
  const float* desc0  = (const float*)d_in[0];
  const float* desc1  = (const float*)d_in[1];
  const float* kpts0  = (const float*)d_in[2];
  const float* kpts1  = (const float*)d_in[3];
  const float* score0 = (const float*)d_in[4];
  const float* score1 = (const float*)d_in[5];
  const float* binsc  = (const float*)d_in[6];
  enum { P_BN_B = 0, P_BN_G, P_BN_M, P_BN_V, P_K_W, P_K_B, P_MERGE_W, P_MERGE_B,
         P_MLP0_W, P_MLP0_B, P_MLP1_W, P_MLP1_B, P_Q_W, P_Q_B, P_V_W, P_V_B };
  auto gp = [&](int l, int f) { return (const float*)d_in[7 + l * 16 + f]; };
  const int KB = 7 + LAYERS * 16;                       // kenc base = 295
  const float* projW = (const float*)d_in[KB + 20];     // 315
  const float* projb = (const float*)d_in[KB + 21];     // 316

  // ---- workspace carve ----
  char* wsp = (char*)d_ws; size_t off = 0;
  auto alloc = [&](size_t bytes) -> void* {
    void* p = wsp + off; off += (bytes + 255) & ~(size_t)255; return p;
  };
  f16* d0h  = (f16*)alloc((size_t)D * N * 2);
  f16* d1h  = (f16*)alloc((size_t)D * N * 2);
  f16* Qh   = (f16*)alloc((size_t)D * N * 2);
  f16* Khb  = (f16*)alloc((size_t)D * N * 2);
  f16* Vh   = (f16*)alloc((size_t)D * N * 2);
  f16* msgh = (f16*)alloc((size_t)D * N * 2);
  f16* hbuf = (f16*)alloc((size_t)H2 * N * 2);
  f16* h2   = (f16*)alloc((size_t)H2 * N * 2);
  f16* dl0  = (f16*)alloc((size_t)D * N * 2);
  f16* dl1  = (f16*)alloc((size_t)D * N * 2);
  f16* m0h  = (f16*)alloc((size_t)D * N * 2);
  f16* m1h  = (f16*)alloc((size_t)D * N * 2);
  f16* ebuf = (f16*)alloc((size_t)5 * N * 2);
  f16* t32  = (f16*)alloc((size_t)32 * N * 2);
  f16* t64  = (f16*)alloc((size_t)64 * N * 2);
  f16* t128 = (f16*)alloc((size_t)128 * N * 2);
  float* S  = (float*)alloc((size_t)N * N * 4);
  float* C  = (float*)alloc((size_t)NB * NB * 4);
  float* u  = (float*)alloc((size_t)NB * 4);
  float* v  = (float*)alloc((size_t)NB * 4);
  const int NSLICE = 16;
  float* pm = (float*)alloc((size_t)NSLICE * NB * 4);
  float* ps = (float*)alloc((size_t)NSLICE * NB * 4);
  float* mx0 = (float*)alloc((size_t)N * 4);
  int*   ix0 = (int*)alloc((size_t)N * 4);
  int*   ix1 = (int*)alloc((size_t)N * 4);
  int*   val0 = (int*)alloc((size_t)N * 4);

  float* Zout     = (float*)d_out;
  float* out_idx0 = Zout + (size_t)NB * NB;
  float* out_idx1 = out_idx0 + N;
  float* out_msc0 = out_idx1 + N;
  float* out_msc1 = out_msc0 + N;

  // ---- keypoint encoder: 5 -> 32 -> 64 -> 128 -> 256, + descᵀ residual ----
  auto run_kenc = [&](const float* kp, const float* sc, const float* desc, f16* dout) {
    build_e_kernel<<<(N + 255) / 256, 256, 0, stream>>>(kp, sc, ebuf, N);
    { Ep e; e.bias = (const float*)d_in[KB + 1];
      e.bn_b = (const float*)d_in[KB + 2]; e.bn_g = (const float*)d_in[KB + 3];
      e.bn_m = (const float*)d_in[KB + 4]; e.bn_v = (const float*)d_in[KB + 5];
      e.relu = 1; e.out16 = t32; e.o16_ms = N; e.o16_ns = 1;
      launch_gemm<float, f16>(stream, (const float*)d_in[KB + 0], 5, 1, ebuf, N, 1, 32, N, 5, e); }
    { Ep e; e.bias = (const float*)d_in[KB + 7];
      e.bn_b = (const float*)d_in[KB + 8]; e.bn_g = (const float*)d_in[KB + 9];
      e.bn_m = (const float*)d_in[KB + 10]; e.bn_v = (const float*)d_in[KB + 11];
      e.relu = 1; e.out16 = t64; e.o16_ms = N; e.o16_ns = 1;
      launch_gemm<float, f16>(stream, (const float*)d_in[KB + 6], 32, 1, t32, N, 1, 64, N, 32, e); }
    { Ep e; e.bias = (const float*)d_in[KB + 13];
      e.bn_b = (const float*)d_in[KB + 14]; e.bn_g = (const float*)d_in[KB + 15];
      e.bn_m = (const float*)d_in[KB + 16]; e.bn_v = (const float*)d_in[KB + 17];
      e.relu = 1; e.out16 = t128; e.o16_ms = N; e.o16_ns = 1;
      launch_gemm<float, f16>(stream, (const float*)d_in[KB + 12], 64, 1, t64, N, 1, 128, N, 64, e); }
    { Ep e; e.bias = (const float*)d_in[KB + 19];
      e.res32 = desc; e.r32_ms = 1; e.r32_ns = D;     // + desc[n][c] (transposed)
      e.out16 = dout; e.o16_ms = N; e.o16_ns = 1;
      launch_gemm<float, f16>(stream, (const float*)d_in[KB + 18], 128, 1, t128, N, 1, 256, N, 128, e); }
  };
  run_kenc(kpts0, score0, desc0, d0h);
  run_kenc(kpts1, score1, desc1, d1h);

  // ---- GNN: 18 alternating self/cross attention layers ----
  auto run_prop = [&](const f16* x, const f16* src, int l, f16* delta) {
    { Ep e; e.bias = gp(l, P_Q_B); e.out16 = Qh; e.o16_ms = N; e.o16_ns = 1;
      launch_gemm<float, f16>(stream, gp(l, P_Q_W), D, 1, x, N, 1, D, N, D, e); }
    { Ep e; e.bias = gp(l, P_K_B); e.out16 = Khb; e.o16_ms = N; e.o16_ns = 1;
      launch_gemm<float, f16>(stream, gp(l, P_K_W), D, 1, src, N, 1, D, N, D, e); }
    { Ep e; e.bias = gp(l, P_V_B); e.out16 = Vh; e.o16_ms = N; e.o16_ns = 1;
      launch_gemm<float, f16>(stream, gp(l, P_V_W), D, 1, src, N, 1, D, N, D, e); }
    for (int h = 0; h < H; ++h) {
      // S[n][m] = (1/8) * sum_d Q[d*4+h][n] * K[d*4+h][m]
      { Ep e; e.alpha = 0.125f; e.out32 = S; e.o32_ms = N; e.o32_ns = 1;
        launch_gemm<f16, f16>(stream, Qh + h * N, 1, (long)H * N,
                              Khb + h * N, (long)H * N, 1, N, N, HD, e); }
      row_softmax_kernel<<<N, 256, 0, stream>>>(S, N);
      // msg[d*4+h][n] = sum_m V[d*4+h][m] * P[n][m]   (B = Pᵀ via strides)
      { Ep e; e.out16 = msgh + h * N; e.o16_ms = (long)H * N; e.o16_ns = 1;
        launch_gemm<f16, float>(stream, Vh + h * N, (long)H * N, 1, S, 1, N, HD, N, N, e); }
    }
    // merge conv -> channels 256..511 of hbuf; x copied to channels 0..255
    { Ep e; e.bias = gp(l, P_MERGE_B); e.out16 = hbuf + (size_t)D * N; e.o16_ms = N; e.o16_ns = 1;
      launch_gemm<float, f16>(stream, gp(l, P_MERGE_W), D, 1, msgh, N, 1, D, N, D, e); }
    hipMemcpyAsync(hbuf, x, (size_t)D * N * sizeof(f16), hipMemcpyDeviceToDevice, stream);
    { Ep e; e.bias = gp(l, P_MLP0_B);
      e.bn_g = gp(l, P_BN_G); e.bn_b = gp(l, P_BN_B);
      e.bn_m = gp(l, P_BN_M); e.bn_v = gp(l, P_BN_V);
      e.relu = 1; e.out16 = h2; e.o16_ms = N; e.o16_ns = 1;
      launch_gemm<float, f16>(stream, gp(l, P_MLP0_W), H2, 1, hbuf, N, 1, H2, N, H2, e); }
    { Ep e; e.bias = gp(l, P_MLP1_B); e.out16 = delta; e.o16_ms = N; e.o16_ns = 1;
      launch_gemm<float, f16>(stream, gp(l, P_MLP1_W), H2, 1, h2, N, 1, D, N, H2, e); }
  };

  for (int l = 0; l < LAYERS; ++l) {
    bool cross = (l & 1) != 0;   // NAMES = ['self','cross']*9
    run_prop(d0h, cross ? d1h : d0h, l, dl0);
    run_prop(d1h, cross ? d0h : d1h, l, dl1);
    resadd_kernel<<<(D * N + 255) / 256, 256, 0, stream>>>(d0h, dl0, D * N);
    resadd_kernel<<<(D * N + 255) / 256, 256, 0, stream>>>(d1h, dl1, D * N);
  }

  // ---- final projection + score matrix into couplings (ld = 2049) ----
  { Ep e; e.bias = projb; e.out16 = m0h; e.o16_ms = N; e.o16_ns = 1;
    launch_gemm<float, f16>(stream, projW, D, 1, d0h, N, 1, D, N, D, e); }
  { Ep e; e.bias = projb; e.out16 = m1h; e.o16_ms = N; e.o16_ns = 1;
    launch_gemm<float, f16>(stream, projW, D, 1, d1h, N, 1, D, N, D, e); }
  { Ep e; e.alpha = 1.0f / 16.0f; e.out32 = C; e.o32_ms = NB; e.o32_ns = 1;
    launch_gemm<f16, f16>(stream, m0h, 1, N, m1h, N, 1, N, N, D, e); }
  fill_bins_kernel<<<(NB + 255) / 256, 256, 0, stream>>>(C, NB, binsc);
  zero_uv_kernel<<<(NB + 255) / 256, 256, 0, stream>>>(u, v, NB);

  // ---- Sinkhorn (100 iterations, log-domain) ----
  const float norm     = -logf((float)(N + N));        // -log(4096)
  const float lmu_last = logf((float)N) + norm;        // = -log 2 (m == n)
  const int   chunk    = (NB + NSLICE - 1) / NSLICE;
  for (int it = 0; it < 100; ++it) {
    sink_row_kernel<<<NB, 256, 0, stream>>>(C, v, u, NB, norm, lmu_last);
    sink_col_part_kernel<<<dim3((NB + 255) / 256, NSLICE), 256, 0, stream>>>(C, u, pm, ps, NB, chunk);
    sink_col_comb_kernel<<<(NB + 255) / 256, 256, 0, stream>>>(pm, ps, v, NB, NSLICE, norm, lmu_last);
  }
  {
    long total = (long)NB * NB;
    write_z_kernel<<<(int)((total + 255) / 256), 256, 0, stream>>>(C, u, v, Zout, NB, -norm);
  }

  // ---- matching ----
  row_argmax_kernel<<<N, 256, 0, stream>>>(Zout, NB, N, mx0, ix0);
  col_argmax_kernel<<<(N + 255) / 256, 256, 0, stream>>>(Zout, NB, N, N, ix1);
  match0_kernel<<<(N + 255) / 256, 256, 0, stream>>>(mx0, ix0, ix1, out_idx0, out_msc0, val0, N);
  match1_kernel<<<(N + 255) / 256, 256, 0, stream>>>(ix0, ix1, out_msc0, val0, out_idx1, out_msc1, N);
}
